// SQN_18743237279796
// MI455X (gfx1250) — matrix operand order repaired and checked by reading, not compiled, not run on hardware
//
#include <hip/hip_runtime.h>
#include <stdint.h>

typedef __attribute__((ext_vector_type(16))) __bf16 v16bf;
typedef __attribute__((ext_vector_type(8)))  float  v8f;
typedef __attribute__((ext_vector_type(2)))  int    v2i;

// ---------------------------------------------------------------------------
// CDNA5 async global->LDS staging (ASYNCcnt path), guarded so the file still
// compiles if this toolchain doesn't expose the gfx1250 async builtins.
// Probe result: params are (v2i addrspace(1)*, v2i addrspace(3)*, imm, imm).
#if __has_builtin(__builtin_amdgcn_global_load_async_to_lds_b64) && \
    __has_builtin(__builtin_amdgcn_s_wait_asynccnt)
#define HAVE_ASYNC_LDS 1
__device__ __forceinline__ void async_copy_b64(const void* g, void* l) {
  __builtin_amdgcn_global_load_async_to_lds_b64(
      (__attribute__((address_space(1))) v2i*)(uintptr_t)g,
      (__attribute__((address_space(3))) v2i*)(uint32_t)(uintptr_t)l,
      0, 0);
}
__device__ __forceinline__ void async_wait0() {
  __builtin_amdgcn_s_wait_asynccnt(0);
}
#else
#define HAVE_ASYNC_LDS 0
#endif

// ---------------------------------------------------------------------------
// WMMA fragment loaders (wave32; same contiguous k = 16*(lane/16)+i order for A and B)
// A: 16x32 bf16. lane<16: row=lane,  K=0..15; lane>=16: row=lane-16,
// K=16..31. At is [rows][32 k] row-major -> 2x16B contiguous LDS reads
__device__ __forceinline__ v16bf load_a_frag(const __bf16* At, int lane, int mtile) {
  int row   = mtile * 16 + (lane & 15);
  int khalf = (lane >> 4) ? 16 : 0;
  const __bf16* p = At + row * 32;
  v16bf a;
#pragma unroll
  for (int i = 0; i < 8; ++i) a[i] = p[khalf + i];
#pragma unroll
  for (int i = 0; i < 8; ++i) a[8 + i] = p[khalf + 8 + i];
  return a;
}

// B: 32x16 bf16 (KxN). lane<16: col=lane, K=0..15; lane>=16: col=lane-16,
// K=16..31. Bt stored [64 n][32 k] -> one 32B contiguous LDS read
__device__ __forceinline__ v16bf load_b_frag(const __bf16* Bt, int lane, int ni) {
  int n    = ni * 16 + (lane & 15);
  int koff = (lane >> 4) ? 16 : 0;
  const __bf16* p = Bt + n * 32 + koff;
  v16bf b;
#pragma unroll
  for (int i = 0; i < 16; ++i) b[i] = p[i];
  return b;
}

// ---------------------------------------------------------------------------
// Implicit-GEMM convolution with WMMA bf16.
//   out[co][n] = relu(bias[co] + sum_k W[co][k]*col[k][n]), n = b*OH*OW+oh*OW+ow
//   input element (b,ci,ih,iw) at in[ci*strideC + b*strideB + ih*IW + iw]
// MT = number of 16-row M subtiles per block (2 or 4). Each of 8 waves owns
// one 16-col N subtile and MT/2 M subtiles -> MT/2 WMMAs per K-step per wave.
// Requires: K%32==0, Cout%(MT*16)==0, N%64==0 (true for all three convs).
template <typename TIN, int MT>
__global__ __launch_bounds__(256) void conv_wmma_kernel(
    const TIN* __restrict__ in, const __bf16* __restrict__ wbf,
    const float* __restrict__ bias, __bf16* __restrict__ out,
    int IH, int IW, int KH, int KW, int stride, int OH, int OW,
    int K, int N, int strideB, int strideC) {
  constexpr int MW = MT / 2;  // M subtiles per wave
  __shared__ __bf16 At[MT * 16 * 32];
  __shared__ __bf16 Bt[64 * 32];

  const int tid    = threadIdx.x;
  const int lane   = tid & 31;
  const int wave   = tid >> 5;
  const int mhalf  = wave & 1;
  const int ni     = wave >> 1;
  const int nBase  = blockIdx.x * 64;
  const int coBase = blockIdx.y * (MT * 16);

  const int kk  = tid & 31;  // fixed k-lane within a K-step
  const int grp = tid >> 5;  // 0..7
  const int OHW = OH * OW;
  const int KHW = KH * KW;

  // Hoist the spatial part of the B gather (n fixed across the K loop)
  int nb_off[8];
#pragma unroll
  for (int e = 0; e < 8; ++e) {
    int ng  = nBase + grp + e * 8;
    int b   = ng / OHW;
    int rem = ng - b * OHW;
    int oh  = rem / OW;
    int ow  = rem - oh * OW;
    nb_off[e] = b * strideB + (oh * stride) * IW + (ow * stride);
  }

  // A-tile async staging geometry: rows of 64B, one b64 chunk per lane/pass
  const int arow = tid >> 3;  // + e*32
  const int achk = tid & 7;

  v8f acc[MW] = {};
  for (int k0 = 0; k0 < K; k0 += 32) {
    const int kg = k0 + kk;
    int ci = kg / KHW;
    int r2 = kg - ci * KHW;
    int kh = r2 / KW;
    int kw = r2 - kh * KW;
    const int in_off = ci * strideC + kh * IW + kw;

    __syncthreads();  // protect LDS from previous iteration's readers
#if HAVE_ASYNC_LDS
#pragma unroll
    for (int e = 0; e < MT / 2; ++e) {  // A: contiguous 8B chunks, async to LDS
      int row = arow + e * 32;
      async_copy_b64(wbf + (coBase + row) * K + k0 + achk * 4,
                     &At[row * 32 + achk * 4]);
    }
#else
#pragma unroll
    for (int e = 0; e < MT; ++e) {
      int row = grp + e * 8;
      At[row * 32 + kk] = wbf[(coBase + row) * K + kg];
    }
#endif
#pragma unroll
    for (int e = 0; e < 8; ++e) {  // B: 32(K)x64(N) im2col tile, stored [n][k]
      int n = grp + e * 8;
      Bt[n * 32 + kk] = (__bf16)(float)in[in_off + nb_off[e]];
    }
#if HAVE_ASYNC_LDS
    async_wait0();  // ASYNCcnt==0 before the barrier that publishes At
#endif
    __syncthreads();

    v16bf b = load_b_frag(Bt, lane, ni);
#pragma unroll
    for (int m = 0; m < MW; ++m) {
      v16bf a = load_a_frag(At, lane, mhalf * MW + m);
      acc[m] = __builtin_amdgcn_wmma_f32_16x16x32_bf16(false, a, false, b,
                                                       (short)0, acc[m],
                                                       false, false);
    }
  }

  // D layout: lane<16 -> rows r, lane>=16 -> rows r+8; col = lane&15
  const int n     = nBase + ni * 16 + (lane & 15);
  const int rbase = (lane >> 4) ? 8 : 0;
#pragma unroll
  for (int m = 0; m < MW; ++m) {
#pragma unroll
    for (int r = 0; r < 8; ++r) {
      int co  = coBase + (mhalf * MW + m) * 16 + rbase + r;
      float v = acc[m][r] + bias[co];
      v = v > 0.0f ? v : 0.0f;
      out[co * N + n] = (__bf16)v;
    }
  }
}

// ---------------------------------------------------------------------------
// fc1: h1[b][j] += sum_k x[b][k]*w1[j][k]  (M=128 batch, N=128 units, K=36954)
// x gathered from conv3 output ([c][b*576+s] bf16) and history (f32).
// Split-K over blockIdx.z with f32 atomic reduction into h1.
#define FC_K      36954
#define FC_KIMG   36864   // 64*24*24, multiple of 32 -> uniform branch
#define FC_S3     576
#define FC_N3     73728   // 128*576
#define FC_KSTEPS 1155    // ceil(36954/32)
#define FC_CHUNK  37      // ksteps per z-chunk -> 32 chunks

__global__ __launch_bounds__(256) void fc1_wmma_kernel(
    const __bf16* __restrict__ out3, const float* __restrict__ history,
    const __bf16* __restrict__ w1bf, float* __restrict__ h1) {
  __shared__ __bf16 At[32 * 32];
  __shared__ __bf16 Bt[64 * 32];

  const int tid   = threadIdx.x;
  const int lane  = tid & 31;
  const int wave  = tid >> 5;
  const int mi    = wave & 1;
  const int ni    = wave >> 1;
  const int nBase = blockIdx.x * 64;  // output unit j
  const int mBase = blockIdx.y * 32;  // batch row b
  const int ks0   = blockIdx.z * FC_CHUNK;
  int ks1 = ks0 + FC_CHUNK;
  if (ks1 > FC_KSTEPS) ks1 = FC_KSTEPS;

  const int kk  = tid & 31;
  const int grp = tid >> 5;

  v8f acc = {};
  for (int ks = ks0; ks < ks1; ++ks) {
    const int k = ks * 32 + kk;
    __syncthreads();
    if (ks < FC_KIMG / 32) {  // scalar-uniform: whole K-step inside the image
      int c = k / FC_S3;
      int s = k - c * FC_S3;
      const __bf16* src = out3 + c * FC_N3 + s;
#pragma unroll
      for (int e = 0; e < 4; ++e) {
        int row = grp + e * 8;
        At[row * 32 + kk] = src[(mBase + row) * FC_S3];
      }
    } else {  // last 3 K-steps: history + zero tail
#pragma unroll
      for (int e = 0; e < 4; ++e) {
        int row = grp + e * 8;
        float v = 0.0f;
        if (k < FC_K && k >= FC_KIMG)
          v = history[(mBase + row) * 90 + (k - FC_KIMG)];
        At[row * 32 + kk] = (__bf16)v;
      }
    }
#pragma unroll
    for (int e = 0; e < 8; ++e) {  // B: w1[j][k] stored [j][k]
      int j = grp + e * 8;
      __bf16 v = (__bf16)0.0f;
      if (k < FC_K) v = w1bf[(nBase + j) * FC_K + k];
      Bt[j * 32 + kk] = v;
    }
    __syncthreads();

    v16bf a = load_a_frag(At, lane, mi);
    v16bf b = load_b_frag(Bt, lane, ni);
    acc = __builtin_amdgcn_wmma_f32_16x16x32_bf16(false, a, false, b,
                                                  (short)0, acc, false, false);
  }

  const int j     = nBase + ni * 16 + (lane & 15);
  const int rbase = (lane >> 4) ? 8 : 0;
#pragma unroll
  for (int r = 0; r < 8; ++r) {
    int b = mBase + mi * 16 + rbase + r;
    atomicAdd(&h1[b * 128 + j], acc[r]);
  }
}

// ---------------------------------------------------------------------------
// LIF recurrence: one workgroup per sample, T=10 steps locally.
// w2 staged once into LDS with stride-129 pad (stride 128 => 32-way conflict).
// Dynamic LDS: 128 + 256 + 256*129 floats = 133,632 B (< 320KB/WGP).
__global__ __launch_bounds__(256) void snn_kernel(
    const float* __restrict__ h1, const float* __restrict__ b1,
    const float* __restrict__ w2, const float* __restrict__ b2,
    const float* __restrict__ w3, const float* __restrict__ b3,
    float* __restrict__ out) {
  extern __shared__ float smem[];
  float* spk1 = smem;        // [128]
  float* spk2 = smem + 128;  // [256]
  float* w2s  = smem + 384;  // [256][129]

  const int b = blockIdx.x;
  const int j = threadIdx.x;

  for (int i = j; i < 256 * 128; i += 256) {  // stage w2
    int row = i >> 7, col = i & 127;
    w2s[row * 129 + col] = w2[i];
  }

  const float AL = 0.9f, BE = 0.8f, TH = 1.0f;
  float mem1 = 0.f, syn1 = 0.f, h1j = 0.f;
  float mem2 = 0.f, syn2 = 0.f;
  float mem3 = 0.f, syn3 = 0.f, pot = 0.f;
  if (j < 128) h1j = h1[b * 128 + j] + b1[j];
  __syncthreads();

  for (int t = 0; t < 10; ++t) {
    if (j < 128) {  // layer-1 LIF (h1 constant across t)
      syn1 = AL * syn1 + h1j;
      mem1 = BE * mem1 + syn1;
      float s = (mem1 - TH > 0.0f) ? 1.0f : 0.0f;
      mem1 -= s * TH;
      spk1[j] = s;
    }
    __syncthreads();
    float h2 = b2[j];  // h2[j] = spk1 . w2[j,:]
#pragma unroll 4
    for (int k = 0; k < 128; ++k) h2 = fmaf(spk1[k], w2s[j * 129 + k], h2);
    syn2 = AL * syn2 + h2;
    mem2 = BE * mem2 + syn2;
    float s2 = (mem2 - TH > 0.0f) ? 1.0f : 0.0f;
    mem2 -= s2 * TH;
    spk2[j] = s2;
    __syncthreads();
    if (j < 9) {  // output integrator
      float h3 = b3[j];
#pragma unroll 4
      for (int k = 0; k < 256; ++k) h3 = fmaf(spk2[k], w3[j * 256 + k], h3);
      syn3 = AL * syn3 + h3;
      mem3 = BE * mem3 + syn3;
      pot += mem3;
    }
  }
  if (j < 9) out[b * 9 + j] = pot * 0.1f;
}

// ---------------------------------------------------------------------------
__global__ void cvt_bf16_kernel(const float* __restrict__ src,
                                __bf16* __restrict__ dst, int n) {
  int i = blockIdx.x * 256 + threadIdx.x;
  if (i < n) dst[i] = (__bf16)src[i];
}

__global__ void zero_f32_kernel(float* __restrict__ p, int n) {
  int i = blockIdx.x * 256 + threadIdx.x;
  if (i < n) p[i] = 0.0f;
}

// ---------------------------------------------------------------------------
extern "C" void kernel_launch(void* const* d_in, const int* in_sizes, int n_in,
                              void* d_out, int out_size, void* d_ws, size_t ws_size,
                              hipStream_t stream) {
  const float* state   = (const float*)d_in[0];   // (128,3,224,224)
  const float* history = (const float*)d_in[1];   // (128,90)
  const float* cw1 = (const float*)d_in[2];  const float* cb1 = (const float*)d_in[3];
  const float* cw2 = (const float*)d_in[4];  const float* cb2 = (const float*)d_in[5];
  const float* cw3 = (const float*)d_in[6];  const float* cb3 = (const float*)d_in[7];
  const float* w1  = (const float*)d_in[8];  const float* b1  = (const float*)d_in[9];
  const float* w2  = (const float*)d_in[10]; const float* b2  = (const float*)d_in[11];
  const float* w3  = (const float*)d_in[12]; const float* b3  = (const float*)d_in[13];
  float* out = (float*)d_out;                     // (128,9)

  // Workspace layout (256B-aligned offsets)
  char* ws = (char*)d_ws;
  __bf16* wbf1 = (__bf16*)(ws + 0);         // 32*192
  __bf16* wbf2 = (__bf16*)(ws + 12288);     // 64*512
  __bf16* wbf3 = (__bf16*)(ws + 77824);     // 64*576
  __bf16* w1bf = (__bf16*)(ws + 151552);    // 128*36954
  __bf16* o1   = (__bf16*)(ws + 9611776);   // [32][387200]
  __bf16* o2   = (__bf16*)(ws + 34392576);  // [64][86528]
  __bf16* o3   = (__bf16*)(ws + 45468160);  // [64][73728]
  float*  h1   = (float*) (ws + 54905344);  // [128][128]

  // Weight conversion f32 -> bf16
  cvt_bf16_kernel<<<(6144 + 255) / 256, 256, 0, stream>>>(cw1, wbf1, 6144);
  cvt_bf16_kernel<<<(32768 + 255) / 256, 256, 0, stream>>>(cw2, wbf2, 32768);
  cvt_bf16_kernel<<<(36864 + 255) / 256, 256, 0, stream>>>(cw3, wbf3, 36864);
  cvt_bf16_kernel<<<(4730112 + 255) / 256, 256, 0, stream>>>(w1, w1bf, 4730112);
  zero_f32_kernel<<<64, 256, 0, stream>>>(h1, 16384);

  // conv1: Cout=32 (MT=2), K=192, N=128*55*55=387200   (input NCHW f32)
  conv_wmma_kernel<float, 2><<<dim3(387200 / 64, 1), 256, 0, stream>>>(
      state, wbf1, cb1, o1, 224, 224, 8, 8, 4, 55, 55, 192, 387200,
      /*strideB=*/3 * 224 * 224, /*strideC=*/224 * 224);
  // conv2: Cout=64 (MT=4), K=512, N=128*26*26=86528    (input [ci][b*3025+..])
  conv_wmma_kernel<__bf16, 4><<<dim3(86528 / 64, 1), 256, 0, stream>>>(
      o1, wbf2, cb2, o2, 55, 55, 4, 4, 2, 26, 26, 512, 86528,
      /*strideB=*/3025, /*strideC=*/387200);
  // conv3: Cout=64 (MT=4), K=576, N=128*24*24=73728
  conv_wmma_kernel<__bf16, 4><<<dim3(73728 / 64, 1), 256, 0, stream>>>(
      o2, wbf3, cb3, o3, 26, 26, 3, 3, 1, 24, 24, 576, 73728,
      /*strideB=*/676, /*strideC=*/86528);

  // fc1 (split-K x32, atomic f32 reduce into h1)
  fc1_wmma_kernel<<<dim3(2, 4, 32), 256, 0, stream>>>(o3, history, w1bf, h1);

  // LIF recurrence, one block per sample; dynamic LDS for padded w2 + spikes
  snn_kernel<<<128, 256, (384 + 256 * 129) * sizeof(float), stream>>>(
      h1, b1, w2, b2, w3, b3, out);
}
